// ShiftReduceEncoder_34076270526787
// MI455X (gfx1250) — matrix operand order repaired and checked by reading, not compiled, not run on hardware
//
#include <hip/hip_runtime.h>
#include <hip/hip_bf16.h>
#include <cstdint>
#include <cstddef>

// ---------------------------------------------------------------------------
// ShiftReduceEncoder forward on MI455X (gfx1250, wave32, WMMA).
// Single persistent 16-wave workgroup; all recurrent state + biases in LDS;
// all matvecs on the matrix pipe via v_wmma_f32_16x16x32_f16 with a
// column-replicated B operand (batch-1 trick), register-blocked 2 row-tiles
// per B fragment. Weights pre-swizzled into WMMA A-fragment order (f16) in
// d_ws by a prologue kernel.
// ---------------------------------------------------------------------------

typedef __attribute__((ext_vector_type(16))) _Float16 v16h;
typedef __attribute__((ext_vector_type(8)))  float    v8f;

constexpr int NTHREADS = 512;
constexpr int NWAVES   = 16;
constexpr int EDIM     = 512;       // embedding dim
constexpr int HDIM     = 512;       // LSTM hidden
constexpr int LAYERS   = 2;
constexpr int GATES    = 4 * HDIM;  // 2048
constexpr int DSTACK   = 4;
constexpr int CATDIM   = 2 * (EDIM + LAYERS * HDIM);  // 3072

// fragment sizes (halves): one 16x32 f16 A-tile = 32 lanes * 16 halves = 512
constexpr size_t FRAG_H      = 512;
constexpr size_t WIH_HALVES  = (size_t)(LAYERS * GATES / 16) * (HDIM / 32) * FRAG_H; // 2,097,152
constexpr size_t WHH_HALVES  = WIH_HALVES;
constexpr size_t WRED_HALVES = (size_t)(EDIM / 16) * (CATDIM / 32) * FRAG_H;         // 1,572,864
constexpr size_t WRES_HALVES = (size_t)(EDIM / 16) * (EDIM / 32) * FRAG_H;           //   262,144

// ---- dynamic LDS layout (bytes) ----
constexpr int OFF_ENC   = 0;       // DSTACK*EDIM f32            8192
constexpr int OFF_HST   = 8192;    // (D+1)*L*H f32             20480
constexpr int OFF_CST   = 28672;   // (D+1)*L*H f32             20480
constexpr int OFF_PRE   = 49152;   // GATES f32                  8192
constexpr int OFF_YBUF  = 57344;   // EDIM f32                   2048
constexpr int OFF_ZX16  = 59392;   // CATDIM f16                 6144
constexpr int OFF_ZH16  = 65536;   // HDIM f16                   1024
constexpr int OFF_RELU  = 66560;   // EDIM f16                   1024
constexpr int OFF_BSUM  = 67584;   // LAYERS*GATES f32 (bih+bhh) 16384
constexpr int OFF_BRED  = 83968;   // EDIM f32                   2048
constexpr int OFF_BRES  = 86016;   // EDIM f32                   2048
constexpr int SMEM_BYTES = 88064;

// ---------------------------------------------------------------------------
// Prologue: swizzle a row-major f32 [R x K] matrix into f16 WMMA A-fragments.
// 16-bit A 16x32 layout used here (contiguous k, B matches): lane L holds row M = L%16; half j maps to
//   K = 16*(L>=16) + j
// Fragment index = tile * (K/32) + kc; each fragment: 32 lanes x 16 halves
// stored lane-contiguous (32 bytes/lane) so the consumer does one v16h load.
// ---------------------------------------------------------------------------
__global__ void swizzle_wmma_f16(const float* __restrict__ W,
                                 _Float16* __restrict__ out,
                                 int R, int K) {
  size_t total = (size_t)(R >> 4) * (size_t)(K >> 5) * FRAG_H;
  size_t idx = (size_t)blockIdx.x * blockDim.x + threadIdx.x;
  if (idx >= total) return;
  int within = (int)(idx & 511);
  size_t frag = idx >> 9;
  int lane = within >> 4;
  int j    = within & 15;
  int kcCount = K >> 5;
  int tile = (int)(frag / kcCount);
  int kc   = (int)(frag % kcCount);
  int row  = tile * 16 + (lane & 15);
  int k    = kc * 32 + ((lane & 16) ? 16 : 0) + j;
  out[idx] = (_Float16)W[(size_t)row * K + k];
}

// ---------------------------------------------------------------------------
__device__ __forceinline__ float sigmoidf_(float x) {
  return 1.0f / (1.0f + __expf(-x));
}

__device__ __forceinline__ v8f wmma_f16(v16h a, v16h b, v8f c) {
  return __builtin_amdgcn_wmma_f32_16x16x32_f16(false, a, false, b, (short)0, c,
                                                false, false);
}

// pre[rows] = swzA @ xA (+ swzB @ xB), raw (biases added later from LDS).
// B operand is column-replicated from the f16 vector in LDS:
//   B[k][n] = x[k]  ->  lane reads 16 contiguous halves at x + 32*kc + (lane&16)
// Register-blocked: 2 row-tiles share each B fragment (two independent WMMA
// accumulation chains -> hides XDL RAW latency, halves DS broadcast traffic).
// With replicated B every lane's D fragment holds valid rows; lanes with
// (lane&15)==0 hold rows (tile*16 + 8*(lane>=16)) + 0..7 and store 32B each.
__device__ void wave_matvec2(float* __restrict__ pre, int rows,
                             const _Float16* __restrict__ swzA,
                             const _Float16* __restrict__ xA, int kcA,
                             const _Float16* __restrict__ swzB,
                             const _Float16* __restrict__ xB, int kcB,
                             int wave, int lane) {
  const int pairs = rows >> 5;        // row-tile pairs
  const int laneSel = lane & 16;      // 0 or 16
  for (int tp = wave; tp < pairs; tp += NWAVES) {
    const int t0 = tp * 2;
    const _Float16* a0 = swzA + (size_t)t0 * kcA * FRAG_H + lane * 16;
    const _Float16* a1 = a0 + (size_t)kcA * FRAG_H;
    if (tp + NWAVES < pairs)
      __builtin_prefetch((const void*)(swzA + (size_t)(tp + NWAVES) * 2 * kcA * FRAG_H), 0, 1);
    v8f c0 = {}, c1 = {};
    for (int kc = 0; kc < kcA; ++kc) {
      v16h b  = *(const v16h*)(xA + kc * 32 + laneSel);
      v16h f0 = *(const v16h*)(a0 + (size_t)kc * FRAG_H);
      v16h f1 = *(const v16h*)(a1 + (size_t)kc * FRAG_H);
      c0 = wmma_f16(f0, b, c0);
      c1 = wmma_f16(f1, b, c1);
    }
    if (swzB != nullptr) {
      const _Float16* b0 = swzB + (size_t)t0 * kcB * FRAG_H + lane * 16;
      const _Float16* b1 = b0 + (size_t)kcB * FRAG_H;
      for (int kc = 0; kc < kcB; ++kc) {
        v16h b  = *(const v16h*)(xB + kc * 32 + laneSel);
        v16h f0 = *(const v16h*)(b0 + (size_t)kc * FRAG_H);
        v16h f1 = *(const v16h*)(b1 + (size_t)kc * FRAG_H);
        c0 = wmma_f16(f0, b, c0);
        c1 = wmma_f16(f1, b, c1);
      }
    }
    if ((lane & 15) == 0) {
      int r0 = t0 * 16 + (laneSel ? 8 : 0);
      *(v8f*)(pre + r0)      = c0;   // rows of tile t0
      *(v8f*)(pre + r0 + 16) = c1;   // rows of tile t0+1
    }
  }
}

// One 2-layer LSTM time-step. Input (f16) already in zx16[0..H); reads
// h_st/c_st slot `src`, writes slot `dst`. PyTorch gate order i,f,g,o.
__device__ void lstm_step(int src, int dst,
                          float* __restrict__ h_st, float* __restrict__ c_st,
                          float* __restrict__ pre, float* __restrict__ bsum,
                          _Float16* __restrict__ zx16, _Float16* __restrict__ zh16,
                          const _Float16* __restrict__ wihS,
                          const _Float16* __restrict__ whhS,
                          int tid, int wave, int lane) {
  constexpr size_t LAYER_STRIDE = (size_t)(GATES / 16) * (HDIM / 32) * FRAG_H; // 1,048,576
  for (int l = 0; l < LAYERS; ++l) {
    zh16[tid] = (_Float16)h_st[(src * LAYERS + l) * HDIM + tid];
    __syncthreads();
    wave_matvec2(pre, GATES,
                 wihS + (size_t)l * LAYER_STRIDE, zx16, HDIM / 32,
                 whhS + (size_t)l * LAYER_STRIDE, zh16, HDIM / 32,
                 wave, lane);
    __syncthreads();
    {
      const float* bl = bsum + l * GATES;
      float gi = pre[tid]            + bl[tid];
      float gf = pre[HDIM + tid]     + bl[HDIM + tid];
      float gg = pre[2 * HDIM + tid] + bl[2 * HDIM + tid];
      float go = pre[3 * HDIM + tid] + bl[3 * HDIM + tid];
      float cp = c_st[(src * LAYERS + l) * HDIM + tid];
      float cn = sigmoidf_(gf) * cp + sigmoidf_(gi) * tanhf(gg);
      float hn = sigmoidf_(go) * tanhf(cn);
      h_st[(dst * LAYERS + l) * HDIM + tid] = hn;
      c_st[(dst * LAYERS + l) * HDIM + tid] = cn;
      zx16[tid] = (_Float16)hn;  // input for next layer
    }
    __syncthreads();
  }
}

// ---------------------------------------------------------------------------
// Persistent single-workgroup kernel: runs all 2047 shift/reduce steps.
// ---------------------------------------------------------------------------
__global__ void __launch_bounds__(NTHREADS, 1)
shift_reduce_kernel(const int* __restrict__ tokens, const int* __restrict__ actions,
                    const float* __restrict__ embed,
                    const float* __restrict__ bih, const float* __restrict__ bhh,
                    const float* __restrict__ h0, const float* __restrict__ c0,
                    const float* __restrict__ bred, const float* __restrict__ bres,
                    const _Float16* __restrict__ wihS, const _Float16* __restrict__ whhS,
                    const _Float16* __restrict__ wredS, const _Float16* __restrict__ wresS,
                    float* __restrict__ out, int steps, int nTok) {
  extern __shared__ v8f smem_v[];   // 32-byte aligned dynamic LDS
  char* smem = (char*)smem_v;
  float*    enc_st = (float*)(smem + OFF_ENC);
  float*    h_st   = (float*)(smem + OFF_HST);
  float*    c_st   = (float*)(smem + OFF_CST);
  float*    pre    = (float*)(smem + OFF_PRE);
  float*    ybuf   = (float*)(smem + OFF_YBUF);
  _Float16* zx16   = (_Float16*)(smem + OFF_ZX16);
  _Float16* zh16   = (_Float16*)(smem + OFF_ZH16);
  _Float16* relu16 = (_Float16*)(smem + OFF_RELU);
  float*    bsum   = (float*)(smem + OFF_BSUM);
  float*    bredL  = (float*)(smem + OFF_BRED);
  float*    bresL  = (float*)(smem + OFF_BRES);

  const int tid  = threadIdx.x;
  const int lane = tid & 31;
  const int wave = tid >> 5;

  // ---- one-time init: zero stacks, load h0/c0, pre-sum biases into LDS ----
  for (int j = tid; j < DSTACK * EDIM; j += NTHREADS) enc_st[j] = 0.0f;
  for (int j = tid; j < (DSTACK + 1) * LAYERS * HDIM; j += NTHREADS) {
    h_st[j] = 0.0f; c_st[j] = 0.0f;
  }
  for (int j = tid; j < LAYERS * GATES; j += NTHREADS) bsum[j] = bih[j] + bhh[j];
  bredL[tid] = bred[tid];
  bresL[tid] = bres[tid];
  __syncthreads();
  if (tid < LAYERS * HDIM) { h_st[tid] = h0[tid]; c_st[tid] = c0[tid]; }
  __syncthreads();

  int p = 0, shifts = 0;
  for (int s = 0; s < steps; ++s) {
    const int act = actions[s];   // block-uniform -> uniform branches, EXEC stays full
    if (act == 0) {
      // ---- SHIFT: x = embed[tokens[bidx]]; LSTM from slot p -> slot p+1 ----
      int bi = shifts < nTok - 1 ? shifts : nTok - 1;
      shifts++;
      int token = tokens[bi];
      float v = embed[(size_t)token * EDIM + tid];
      enc_st[p * EDIM + tid] = v;
      zx16[tid] = (_Float16)v;
      __syncthreads();
      lstm_step(p, p + 1, h_st, c_st, pre, bsum, zx16, zh16, wihS, whhS,
                tid, wave, lane);
      p += 1;
    } else {
      // ---- REDUCE: cat = [c_st[p], enc[p-1], c_st[p-1], enc[p-2]] (3072) ----
      zx16[tid]                    = (_Float16)c_st[(p * LAYERS + 0) * HDIM + tid];
      zx16[HDIM + tid]             = (_Float16)c_st[(p * LAYERS + 1) * HDIM + tid];
      zx16[2 * HDIM + tid]         = (_Float16)enc_st[(p - 1) * EDIM + tid];
      zx16[2 * HDIM + EDIM + tid]  = (_Float16)c_st[((p - 1) * LAYERS + 0) * HDIM + tid];
      zx16[3 * HDIM + EDIM + tid]  = (_Float16)c_st[((p - 1) * LAYERS + 1) * HDIM + tid];
      zx16[4 * HDIM + EDIM + tid]  = (_Float16)enc_st[(p - 2) * EDIM + tid];
      __syncthreads();
      // y = Wred @ cat (+ bred below)
      wave_matvec2(pre, EDIM, wredS, zx16, CATDIM / 32,
                   nullptr, nullptr, 0, wave, lane);
      __syncthreads();
      { float y = pre[tid] + bredL[tid];
        ybuf[tid] = y;
        relu16[tid] = (_Float16)fmaxf(y, 0.0f); }
      __syncthreads();
      // y = Wres @ relu(y) + bres + y   (ResLayer)
      wave_matvec2(pre, EDIM, wresS, relu16, EDIM / 32,
                   nullptr, nullptr, 0, wave, lane);
      __syncthreads();
      { float yf = pre[tid] + bresL[tid] + ybuf[tid];
        enc_st[(p - 2) * EDIM + tid] = yf;
        zx16[tid] = (_Float16)yf; }
      __syncthreads();
      lstm_step(p - 2, p - 1, h_st, c_st, pre, bsum, zx16, zh16, wihS, whhS,
                tid, wave, lane);
      p -= 1;
    }
  }
  out[tid] = enc_st[tid];  // enc_st[0] : final root encoding [EDIM]
}

// ---------------------------------------------------------------------------
extern "C" void kernel_launch(void* const* d_in, const int* in_sizes, int n_in,
                              void* d_out, int out_size, void* d_ws, size_t ws_size,
                              hipStream_t stream) {
  // setup_inputs() order:
  // 0 input_indices(i32) 1 actions(i32) 2 embed 3 Wih 4 Whh 5 bih 6 bhh
  // 7 h0 8 c0 9 Wact 10 bact 11 Wred 12 bred 13 Wres 14 bres
  const int*   tokens  = (const int*)d_in[0];
  const int*   actions = (const int*)d_in[1];
  const float* embed   = (const float*)d_in[2];
  const float* Wih     = (const float*)d_in[3];
  const float* Whh     = (const float*)d_in[4];
  const float* bih     = (const float*)d_in[5];
  const float* bhh     = (const float*)d_in[6];
  const float* h0      = (const float*)d_in[7];
  const float* c0      = (const float*)d_in[8];
  // Wact/bact (d_in[9], d_in[10]) only affect the straight-through score,
  // which is identity in the forward pass -> not needed for the output.
  const float* Wred    = (const float*)d_in[11];
  const float* bred    = (const float*)d_in[12];
  const float* Wres    = (const float*)d_in[13];
  const float* bres    = (const float*)d_in[14];

  _Float16* wihS  = (_Float16*)d_ws;
  _Float16* whhS  = wihS  + WIH_HALVES;
  _Float16* wredS = whhS  + WHH_HALVES;
  _Float16* wresS = wredS + WRED_HALVES;
  // total ws use: ~11.5 MB of f16 fragments

  const int steps = in_sizes[1];   // 2*N-1 = 2047
  const int nTok  = in_sizes[0];   // N = 1024

  // Prologue: swizzle weights to WMMA fragment order (deterministic, re-run
  // on every call so graph replay stays valid even if ws is repoisoned).
  {
    unsigned b = (unsigned)((WIH_HALVES + 255) / 256);
    swizzle_wmma_f16<<<b, 256, 0, stream>>>(Wih, wihS, LAYERS * GATES, HDIM);
    swizzle_wmma_f16<<<b, 256, 0, stream>>>(Whh, whhS, LAYERS * GATES, HDIM);
  }
  {
    unsigned b = (unsigned)((WRED_HALVES + 255) / 256);
    swizzle_wmma_f16<<<b, 256, 0, stream>>>(Wred, wredS, EDIM, CATDIM);
  }
  {
    unsigned b = (unsigned)((WRES_HALVES + 255) / 256);
    swizzle_wmma_f16<<<b, 256, 0, stream>>>(Wres, wresS, EDIM, EDIM);
  }

  // Persistent single-WGP sequential kernel.
  shift_reduce_kernel<<<1, NTHREADS, SMEM_BYTES, stream>>>(
      tokens, actions, embed, bih, bhh, h0, c0, bred, bres,
      wihS, whhS, wredS, wresS, (float*)d_out, steps, nTok);
}
